// _WindowAttention_128849019425
// MI455X (gfx1250) — matrix operand order repaired and checked by reading, not compiled, not run on hardware
//
#include <hip/hip_runtime.h>
#include <hip/hip_bf16.h>

typedef __attribute__((ext_vector_type(16))) _Float16 v16h;
typedef __attribute__((ext_vector_type(8)))  float    v8f;

#define NTOK   64
#define CDIM   256
#define NHEAD  8
#define HD     32
#define XSTR   264   // padded f16 stride for x / sO rows (64 banks clean)
#define QKSTR  33    // per-wave scratch stride for Q/K/V staging
#define PSTR   66    // per-wave scratch stride for P staging

#define WMMA(A, B, C) \
  __builtin_amdgcn_wmma_f32_16x16x32_f16(false, (A), false, (B), (short)0, (C), false, false)

// K index inside a 16x32 f16 A-fragment for half-register j (0..15), lane-half lh.
// Contiguous: lanes 0..15 hold K in [0,16), lanes 16..31 hold K in [16,32).
__device__ __forceinline__ int kmap(int j, int lh) {
  return lh * 16 + j;
}

__device__ __forceinline__ v16h load_a_frag(const _Float16* __restrict__ base, int stride,
                                            int row0, int col0, int ln, int lh) {
  v16h a;
  const _Float16* p = base + (row0 + ln) * stride + col0;
#pragma unroll
  for (int j = 0; j < 16; ++j) a[j] = p[kmap(j, lh)];
  return a;
}

__device__ __forceinline__ v16h load_b_contig(const _Float16* __restrict__ p) {
  v16h b;
#pragma unroll
  for (int j = 0; j < 16; ++j) b[j] = p[j];
  return b;
}

// ---- prep: f32->f16 weights + relative-position bias table (shared by all windows) ----
__global__ void __launch_bounds__(256) prep_kernel(
    const float* __restrict__ qkv_w, const float* __restrict__ proj_w,
    const float* __restrict__ rpb,  const int* __restrict__ rel_idx,
    _Float16* __restrict__ wqkv, _Float16* __restrict__ wproj,
    float* __restrict__ biasW) {
  int t = blockIdx.x * 256 + threadIdx.x;
  if (t < 768 * 256) wqkv[t] = (_Float16)qkv_w[t];
  int t2 = t - 768 * 256;
  if (t2 >= 0 && t2 < 256 * 256) wproj[t2] = (_Float16)proj_w[t2];
  int t3 = t2 - 256 * 256;
  if (t3 >= 0 && t3 < NHEAD * NTOK * NTOK) {
    int h = t3 >> 12, rem = t3 & 4095;
    biasW[h * 4096 + rem] = rpb[rel_idx[rem] * NHEAD + h];
  }
}

// ---- fused window attention: one block per window, one wave per head ----
__global__ void __launch_bounds__(256) winattn_kernel(
    const float* __restrict__ x, const float* __restrict__ mask,
    const float* __restrict__ qkv_b, const float* __restrict__ proj_b,
    const _Float16* __restrict__ wqkv, const _Float16* __restrict__ wproj,
    const float* __restrict__ biasW, float* __restrict__ out) {
  __shared__ _Float16 sX[NTOK * XSTR];       // x (f16) during QKV; reused as sO after
  __shared__ _Float16 sScr[NHEAD][1056];     // per-wave fragment-transpose scratch

  const int b    = blockIdx.x;
  const int tid  = threadIdx.x;
  const int wv   = tid >> 5;    // wave id == head id
  const int lane = tid & 31;
  const int lh   = lane >> 4;
  const int ln   = lane & 15;

  // ---- stage x -> LDS (f16) ----
  const float* xg = x + (size_t)b * (NTOK * CDIM);
  for (int t = tid; t < NTOK * CDIM; t += 256) {
    int r = t >> 8, c = t & 255;
    sX[r * XSTR + c] = (_Float16)xg[t];
  }
  __syncthreads();

  _Float16* scr = sScr[wv];
  const float scale = 0.17677669529663689f;  // hd^-0.5

  v16h aQ[4];   // Q row-tiles as A fragments (K = hd = 32: single fragment each)
  v16h bK[4];   // K^T as B fragments (N = 16 keys each)
  v16h bV[4];   // V as B fragments: [ks(0,1)][nt(0,1)] -> bV[ks*2+nt]

  // ---- Q and K head slices: [64,32] = x[64,256] @ w_slice^T ----
#pragma unroll
  for (int mat = 0; mat < 2; ++mat) {
#pragma unroll
    for (int mt = 0; mt < 4; ++mt) {
      v8f z = {};
      v8f acc0 = z, acc1 = z;
#pragma unroll
      for (int kk = 0; kk < 8; ++kk) {
        v16h a = load_a_frag(sX, XSTR, mt * 16, kk * 32, ln, lh);
        const _Float16* w0 =
            wqkv + (size_t)(mat * 256 + wv * 32 + ln) * 256 + kk * 32 + lh * 16;
        v16h b0 = load_b_contig(w0);
        v16h b1 = load_b_contig(w0 + 16 * 256);
        acc0 = WMMA(a, b0, acc0);
        acc1 = WMMA(a, b1, acc1);
      }
      // bias (+scale for Q), stage as f16 [16][QKSTR]
#pragma unroll
      for (int n = 0; n < 2; ++n) {
        const v8f acc = n ? acc1 : acc0;
        int col = n * 16 + ln;
        float bias = qkv_b[mat * 256 + wv * 32 + col];
#pragma unroll
        for (int r = 0; r < 8; ++r) {
          float v = acc[r] + bias;
          if (mat == 0) v *= scale;
          scr[(r + 8 * lh) * QKSTR + col] = (_Float16)v;
        }
      }
      if (mat == 0)
        aQ[mt] = load_a_frag(scr, QKSTR, 0, 0, ln, lh);
      else
        bK[mt] = load_b_contig(scr + ln * QKSTR + lh * 16);
    }
  }

  // ---- V head slice, staged transposed [hd][tok] per 32-token chunk ----
#pragma unroll
  for (int ks = 0; ks < 2; ++ks) {
    v8f z = {};
    v8f a00 = z, a01 = z, a10 = z, a11 = z;
#pragma unroll
    for (int kk = 0; kk < 8; ++kk) {
      v16h x0 = load_a_frag(sX, XSTR, (ks * 2) * 16, kk * 32, ln, lh);
      v16h x1 = load_a_frag(sX, XSTR, (ks * 2 + 1) * 16, kk * 32, ln, lh);
      const _Float16* w0 = wqkv + (size_t)(512 + wv * 32 + ln) * 256 + kk * 32 + lh * 16;
      v16h b0 = load_b_contig(w0);
      v16h b1 = load_b_contig(w0 + 16 * 256);
      a00 = WMMA(x0, b0, a00);
      a01 = WMMA(x0, b1, a01);
      a10 = WMMA(x1, b0, a10);
      a11 = WMMA(x1, b1, a11);
    }
#pragma unroll
    for (int i = 0; i < 2; ++i) {
#pragma unroll
      for (int n = 0; n < 2; ++n) {
        const v8f acc = i ? (n ? a11 : a10) : (n ? a01 : a00);
        int col = n * 16 + ln;
        float bias = qkv_b[512 + wv * 32 + col];
#pragma unroll
        for (int r = 0; r < 8; ++r) {
          int tok = i * 16 + r + 8 * lh;
          scr[col * QKSTR + tok] = (_Float16)(acc[r] + bias);  // transposed [32][QKSTR]
        }
      }
    }
#pragma unroll
    for (int nt = 0; nt < 2; ++nt)
      bV[ks * 2 + nt] = load_b_contig(scr + (nt * 16 + ln) * QKSTR + lh * 16);
  }

  __syncthreads();  // all waves done reading sX -> safe to reuse as sO

  // ---- attention: S = Q K^T (+bias+mask), softmax, O = P V ----
  const float* bw = biasW + wv * 4096;
  const float* mk = mask + (size_t)b * 4096;
#pragma unroll
  for (int mi = 0; mi < 4; ++mi) {
    v8f z = {};
    v8f s[4];
#pragma unroll
    for (int ni = 0; ni < 4; ++ni) s[ni] = WMMA(aQ[mi], bK[ni], z);

#pragma unroll
    for (int ni = 0; ni < 4; ++ni) {
      int col = ni * 16 + ln;
#pragma unroll
      for (int r = 0; r < 8; ++r) {
        int row = mi * 16 + r + 8 * lh;
        s[ni][r] += bw[row * 64 + col] + mk[row * 64 + col];
      }
    }
    // row softmax: each row lives in one 16-lane half; xor masks stay in-half
#pragma unroll
    for (int r = 0; r < 8; ++r) {
      float v0 = s[0][r], v1 = s[1][r], v2 = s[2][r], v3 = s[3][r];
      float mx = fmaxf(fmaxf(v0, v1), fmaxf(v2, v3));
#pragma unroll
      for (int off = 8; off >= 1; off >>= 1) mx = fmaxf(mx, __shfl_xor(mx, off, 32));
      v0 = __expf(v0 - mx); v1 = __expf(v1 - mx);
      v2 = __expf(v2 - mx); v3 = __expf(v3 - mx);
      float sum = v0 + v1 + v2 + v3;
#pragma unroll
      for (int off = 8; off >= 1; off >>= 1) sum += __shfl_xor(sum, off, 32);
      float inv = 1.0f / sum;
      s[0][r] = v0 * inv; s[1][r] = v1 * inv; s[2][r] = v2 * inv; s[3][r] = v3 * inv;
    }
    // stage P (f16) and reload as A fragments (2 K-chunks of 32)
#pragma unroll
    for (int ni = 0; ni < 4; ++ni)
#pragma unroll
      for (int r = 0; r < 8; ++r)
        scr[(r + 8 * lh) * PSTR + ni * 16 + ln] = (_Float16)s[ni][r];
    v16h p0 = load_a_frag(scr, PSTR, 0, 0, ln, lh);
    v16h p1 = load_a_frag(scr, PSTR, 0, 32, ln, lh);

    v8f o0 = z, o1 = z;
    o0 = WMMA(p0, bV[0], o0); o0 = WMMA(p1, bV[2], o0);
    o1 = WMMA(p0, bV[1], o1); o1 = WMMA(p1, bV[3], o1);

#pragma unroll
    for (int n = 0; n < 2; ++n) {
      const v8f o = n ? o1 : o0;
#pragma unroll
      for (int r = 0; r < 8; ++r) {
        int row = mi * 16 + r + 8 * lh;
        sX[row * XSTR + wv * 32 + n * 16 + ln] = (_Float16)o[r];  // sO[row][head*32+d]
      }
    }
  }
  __syncthreads();  // sO complete

  // ---- output projection: out[64,256] = sO @ proj_w^T + proj_b ----
#pragma unroll
  for (int t8 = 0; t8 < 8; ++t8) {
    int t = wv * 8 + t8;
    int m = t >> 4, n = t & 15;
    v8f acc = {};
#pragma unroll
    for (int kk = 0; kk < 8; ++kk) {
      v16h a = load_a_frag(sX, XSTR, m * 16, kk * 32, ln, lh);
      const _Float16* wp = wproj + (size_t)(n * 16 + ln) * 256 + kk * 32 + lh * 16;
      v16h bb = load_b_contig(wp);
      acc = WMMA(a, bb, acc);
    }
    int col = n * 16 + ln;
    float pb = proj_b[col];
    float* og = out + ((size_t)b * 64 + m * 16) * 256;
#pragma unroll
    for (int r = 0; r < 8; ++r) {
      int row = r + 8 * lh;
      og[(size_t)row * 256 + col] = acc[r] + pb;
    }
  }
}

extern "C" void kernel_launch(void* const* d_in, const int* in_sizes, int n_in,
                              void* d_out, int out_size, void* d_ws, size_t ws_size,
                              hipStream_t stream) {
  const float* x      = (const float*)d_in[0];
  const float* mask   = (const float*)d_in[1];
  const float* rpb    = (const float*)d_in[2];
  const float* qkv_w  = (const float*)d_in[3];
  const float* qkv_b  = (const float*)d_in[4];
  const float* proj_w = (const float*)d_in[5];
  const float* proj_b = (const float*)d_in[6];
  const int*   relidx = (const int*)d_in[7];

  char* ws = (char*)d_ws;
  _Float16* wqkv  = (_Float16*)ws;                             // 768*256 f16 = 393216 B
  _Float16* wproj = (_Float16*)(ws + 768 * 256 * 2);           // 256*256 f16 = 131072 B
  float*    biasW = (float*)(ws + 768 * 256 * 2 + 256 * 256 * 2);  // 8*64*64 f32 = 131072 B

  const int prep_items = 768 * 256 + 256 * 256 + NHEAD * NTOK * NTOK;
  prep_kernel<<<(prep_items + 255) / 256, 256, 0, stream>>>(
      qkv_w, proj_w, rpb, relidx, wqkv, wproj, biasW);

  winattn_kernel<<<4096, 256, 0, stream>>>(
      x, mask, qkv_b, proj_b, wqkv, wproj, biasW, (float*)d_out);
}
